// GeometryAwarePluckerAttention_63617055588435
// MI455X (gfx1250) — matrix operand order repaired and checked by reading, not compiled, not run on hardware
//
#include <hip/hip_runtime.h>
#include <hip/hip_bf16.h>
#include <math.h>

// ---------------------------------------------------------------------------
// GeometryAwarePluckerAttention for MI455X (gfx1250, wave32, WMMA bf16)
// WMMA fragments are fed by contiguous 16B loads (ds_load_b128 from LDS; Q from global).
// ---------------------------------------------------------------------------

#define B_   4
#define N_   1024
#define C_   1024
#define NH_  16
#define HD_  64
#define PP_  128
#define M_   (B_ * N_)

typedef __attribute__((ext_vector_type(16))) __bf16 v16bf;
typedef __attribute__((ext_vector_type(8)))  float  v8f;

struct __align__(16) Q4 { unsigned int x, y, z, w; };
struct __align__(16) F4 { float x, y, z, w; };

union BF16x16 { v16bf v; Q4 q[2]; unsigned short u[16]; };

__device__ __forceinline__ unsigned short f2bf(float x) {
    unsigned int u = __float_as_uint(x);
    u += 0x7fffu + ((u >> 16) & 1u);          // RNE
    return (unsigned short)(u >> 16);
}
__device__ __forceinline__ unsigned int f2bf2(float a, float b) {
    return (unsigned int)f2bf(a) | ((unsigned int)f2bf(b) << 16);
}
__device__ __forceinline__ float gelu_exact(float x) {
    return 0.5f * x * (1.0f + erff(x * 0.70710678118654752f));
}

// ---------------------------------------------------------------------------
// Tiled WMMA GEMM: Out[M,N] = act(A @ W + bias) + resid.
// Block tile 128x64, 8 waves, each wave 16 rows x 64 cols (4 WMMA / K-step).
// A staged row-major [128][40] (80B stride, 16B aligned);
// W staged transposed [n][k] = [64][40] so B-fragments are contiguous.
// Requires M%128==0, N%64==0, K%32==0.
// ---------------------------------------------------------------------------
__global__ __launch_bounds__(256)
void wmma_gemm_kernel(const float* __restrict__ A, const float* __restrict__ W,
                      const float* __restrict__ bias, const float* __restrict__ resid,
                      float* __restrict__ Out, int M, int K, int N, int act)
{
    __shared__ alignas(16) unsigned short As[128][40];
    __shared__ alignas(16) unsigned short Bt[64][40];

    const int tid  = threadIdx.x;
    const int lane = tid & 31;
    const int wave = tid >> 5;
    const int g    = lane >> 4;
    const int l15  = lane & 15;

    const int nTn = N >> 6;
    const int bm  = (blockIdx.x / nTn) << 7;
    const int bn  = (blockIdx.x % nTn) << 6;

    v8f acc[4];
    { v8f z = {}; for (int i = 0; i < 4; ++i) acc[i] = z; }

    for (int k0 = 0; k0 < K; k0 += 32) {
        // stage A: 128x32 = 2048 bf16-pairs, 8 per thread (contiguous pair loads)
        #pragma unroll
        for (int i = 0; i < 8; ++i) {
            int e = tid + i * 256;
            int r = e >> 4, c = (e & 15) << 1;
            const float* src = &A[(size_t)(bm + r) * K + (k0 + c)];
            *(unsigned int*)&As[r][c] = f2bf2(src[0], src[1]);
        }
        // stage W transposed: 64(n) x 32(k), packed along k
        #pragma unroll
        for (int i = 0; i < 4; ++i) {
            int e = tid + i * 256;
            int c = e & 63, r = (e >> 6) << 1;
            float w0 = W[(size_t)(k0 + r)     * N + (bn + c)];
            float w1 = W[(size_t)(k0 + r + 1) * N + (bn + c)];
            *(unsigned int*)&Bt[c][r] = f2bf2(w0, w1);
        }
        if (k0 + 32 < K)
            __builtin_prefetch(&A[(size_t)(bm + (tid >> 1)) * K + (k0 + 32)], 0, 1);
        __syncthreads();

        BF16x16 a;
        {   // A-frag: K runs {8g..8g+7} and {16+8g..+7} -> two b128
            const Q4* pa = (const Q4*)&As[(wave << 4) + l15][0];
            a.q[0] = pa[g];
            a.q[1] = pa[2 + g];
        }
        #pragma unroll
        for (int j = 0; j < 4; ++j) {
            BF16x16 b;  // B-frag: K runs {8g..8g+7} and {16+8g..+7} -> two b128 from Bt row
            const Q4* pb = (const Q4*)&Bt[(j << 4) + l15][0];
            b.q[0] = pb[g];
            b.q[1] = pb[2 + g];
            acc[j] = __builtin_amdgcn_wmma_f32_16x16x32_bf16(false, a.v, false, b.v,
                                                             (short)0, acc[j], false, false);
        }
        __syncthreads();
    }

    #pragma unroll
    for (int j = 0; j < 4; ++j)
        #pragma unroll
        for (int v = 0; v < 8; ++v) {
            int row = bm + (wave << 4) + v + (g << 3);
            int col = bn + (j << 4) + l15;
            float x0 = acc[j][v];
            if (bias) x0 += bias[col];
            if (act == 1)      x0 = gelu_exact(x0);
            else if (act == 2) x0 = tanhf(x0);
            size_t o = (size_t)row * N + col;
            if (resid) x0 += resid[o];
            Out[o] = x0;
        }
}

// ---------------------------------------------------------------------------
// Flash attention (wave32 WMMA). Block = 128 query rows of one (b,h).
// K staged row-major [key][d] (natural), V staged transposed [d][key]:
// both give contiguous 16B fragment loads.
// ---------------------------------------------------------------------------
template <int CROSS>
__global__ __launch_bounds__(256)
void attn_kernel(const float* __restrict__ Qp, const float* __restrict__ Kp,
                 const float* __restrict__ Vp, int qs, int ks, int vs,
                 const float* __restrict__ rel_table,
                 const float* __restrict__ dn, const float* __restrict__ dsArr,
                 const float* __restrict__ gbArr, const float* __restrict__ rayb,
                 float* __restrict__ Out)
{
    __shared__ alignas(16) unsigned short Kt[64][72];     // [key][d]
    __shared__ alignas(16) unsigned short Vt[64][72];     // [d][key]
    __shared__ alignas(16) unsigned short pS[8][16][72];  // per-wave P: [row][key]

    const int tid  = threadIdx.x;
    const int lane = tid & 31;
    const int wave = tid >> 5;
    const int g    = lane >> 4;
    const int l15  = lane & 15;

    const int nqb = N_ / 128;
    int bid = blockIdx.x;
    const int b  = bid / (NH_ * nqb);  bid -= b * NH_ * nqb;
    const int h  = bid / nqb;
    const int qb = bid % nqb;
    const int q0 = qb * 128 + wave * 16;

    const float scale = 0.125f;

    // Q fragments via 16B global loads
    BF16x16 aq[2];
    {
        const float* qptr = Qp + ((size_t)(b * N_ + q0 + l15)) * qs + h * HD_;
        #pragma unroll
        for (int kk = 0; kk < 2; ++kk) {
            const F4* f = (const F4*)(qptr + kk * 32);
            F4 x0 = f[2 * g], x1 = f[2 * g + 1];       // K 8g..8g+7
            F4 x2 = f[4 + 2 * g], x3 = f[5 + 2 * g];   // K 16+8g..+7
            aq[kk].q[0].x = f2bf2(x0.x, x0.y); aq[kk].q[0].y = f2bf2(x0.z, x0.w);
            aq[kk].q[0].z = f2bf2(x1.x, x1.y); aq[kk].q[0].w = f2bf2(x1.z, x1.w);
            aq[kk].q[1].x = f2bf2(x2.x, x2.y); aq[kk].q[1].y = f2bf2(x2.z, x2.w);
            aq[kk].q[1].z = f2bf2(x3.x, x3.y); aq[kk].q[1].w = f2bf2(x3.z, x3.w);
        }
    }

    float rmax[8], rsum[8];
    v8f acc[4];
    {
        v8f z = {};
        #pragma unroll
        for (int i = 0; i < 4; ++i) acc[i] = z;
        #pragma unroll
        for (int v = 0; v < 8; ++v) { rmax[v] = -1e30f; rsum[v] = 0.0f; }
    }

    float dqx[8], dqy[8], dqz[8], dsq[8];
    float rb = 0.0f;
    if (CROSS) {
        rb = rayb[h];
        #pragma unroll
        for (int v = 0; v < 8; ++v) {
            int qrow = q0 + v + (g << 3);
            const float* dp = dn + (size_t)(b * N_ + qrow) * 3;
            dqx[v] = dp[0]; dqy[v] = dp[1]; dqz[v] = dp[2];
            dsq[v] = 1.0f + dsArr[(size_t)(b * N_ + qrow) * NH_ + h];
        }
    }

    for (int kb = 0; kb < N_; kb += 64) {
        __syncthreads();
        // K tile: natural row-major copy, packed pairs along d
        #pragma unroll
        for (int i = 0; i < 8; ++i) {
            int e = tid + i * 256;
            int m = e >> 5, d = (e & 31) << 1;
            const float* src = Kp + ((size_t)(b * N_ + kb + m)) * ks + h * HD_ + d;
            *(unsigned int*)&Kt[m][d] = f2bf2(src[0], src[1]);
        }
        // V tile: transposed store [d][key], packed pairs along key
        #pragma unroll
        for (int i = 0; i < 8; ++i) {
            int e = tid + i * 256;
            int d = e & 63, m = (e >> 6) << 1;
            float v0 = Vp[((size_t)(b * N_ + kb + m))     * vs + h * HD_ + d];
            float v1 = Vp[((size_t)(b * N_ + kb + m + 1)) * vs + h * HD_ + d];
            *(unsigned int*)&Vt[d][m] = f2bf2(v0, v1);
        }
        __syncthreads();

        // S = scale * Q K^T (+ biases)
        float s[4][8];
        #pragma unroll
        for (int j = 0; j < 4; ++j) {
            v8f c = {};
            #pragma unroll
            for (int kk = 0; kk < 2; ++kk) {
                BF16x16 bf;
                const Q4* pb = (const Q4*)&Kt[(j << 4) + l15][0];
                bf.q[0] = pb[4 * kk + g];
                bf.q[1] = pb[4 * kk + 2 + g];
                c = __builtin_amdgcn_wmma_f32_16x16x32_bf16(false, aq[kk].v, false, bf.v,
                                                            (short)0, c, false, false);
            }
            const int mkey = kb + (j << 4) + l15;
            if (CROSS) {
                const float* dkp = dn + (size_t)(b * N_ + mkey) * 3;
                float dkx = dkp[0], dky = dkp[1], dkz = dkp[2];
                float gbk = gbArr[(size_t)(b * N_ + mkey) * NH_ + h];
                #pragma unroll
                for (int v = 0; v < 8; ++v) {
                    float sim = dqx[v] * dkx + dqy[v] * dky + dqz[v] * dkz;
                    s[j][v] = (c[v] * scale + sim * rb) * dsq[v] + gbk;
                }
            } else {
                int ky = mkey >> 5, kx = mkey & 31;
                #pragma unroll
                for (int v = 0; v < 8; ++v) {
                    int qrow = q0 + v + (g << 3);
                    int idx  = ((qrow >> 5) - ky + 31) * 63 + ((qrow & 31) - kx + 31);
                    s[j][v]  = c[v] * scale + rel_table[(size_t)idx * NH_ + h];
                }
            }
        }

        // online softmax (reduce across the 16 lanes of this half)
        #pragma unroll
        for (int v = 0; v < 8; ++v) {
            float m = fmaxf(fmaxf(s[0][v], s[1][v]), fmaxf(s[2][v], s[3][v]));
            #pragma unroll
            for (int off = 1; off < 16; off <<= 1)
                m = fmaxf(m, __shfl_xor(m, off, 32));
            float mn   = fmaxf(rmax[v], m);
            float corr = __expf(rmax[v] - mn);
            rmax[v] = mn;
            float part = 0.0f;
            #pragma unroll
            for (int j = 0; j < 4; ++j) {
                float p = __expf(s[j][v] - mn);
                s[j][v] = p;
                part += p;
            }
            #pragma unroll
            for (int off = 1; off < 16; off <<= 1)
                part += __shfl_xor(part, off, 32);
            rsum[v] = rsum[v] * corr + part;
            #pragma unroll
            for (int jd = 0; jd < 4; ++jd)
                acc[jd][v] *= corr;
        }

        // stage P for A-fragment reload
        #pragma unroll
        for (int j = 0; j < 4; ++j)
            #pragma unroll
            for (int v = 0; v < 8; ++v)
                pS[wave][v + (g << 3)][(j << 4) + l15] = f2bf(s[j][v]);
        __syncthreads();

        // acc += P @ V
        BF16x16 ap[2];
        {
            const Q4* pp = (const Q4*)&pS[wave][l15][0];
            #pragma unroll
            for (int kk = 0; kk < 2; ++kk) {
                ap[kk].q[0] = pp[4 * kk + g];
                ap[kk].q[1] = pp[4 * kk + 2 + g];
            }
        }
        #pragma unroll
        for (int jd = 0; jd < 4; ++jd) {
            v8f c = acc[jd];
            #pragma unroll
            for (int kk = 0; kk < 2; ++kk) {
                BF16x16 bf;
                const Q4* pb = (const Q4*)&Vt[(jd << 4) + l15][0];
                bf.q[0] = pb[4 * kk + g];
                bf.q[1] = pb[4 * kk + 2 + g];
                c = __builtin_amdgcn_wmma_f32_16x16x32_bf16(false, ap[kk].v, false, bf.v,
                                                            (short)0, c, false, false);
            }
            acc[jd] = c;
        }
    }

    #pragma unroll
    for (int jd = 0; jd < 4; ++jd)
        #pragma unroll
        for (int v = 0; v < 8; ++v) {
            int qrow = q0 + v + (g << 3);
            Out[((size_t)(b * N_ + qrow)) * C_ + h * HD_ + (jd << 4) + l15] =
                acc[jd][v] / rsum[v];
        }
}

// ---------------------------------------------------------------------------
// Scalar GEMM for tiny K / N
// ---------------------------------------------------------------------------
__global__ void small_gemm_kernel(const float* __restrict__ A, const float* __restrict__ W,
                                  const float* __restrict__ bias, float* __restrict__ Out,
                                  int M, int K, int N, int act)
{
    int i = blockIdx.x * blockDim.x + threadIdx.x;
    if (i >= M * N) return;
    int m = i / N, n = i % N;
    float s = bias ? bias[n] : 0.0f;
    for (int k = 0; k < K; ++k) s += A[(size_t)m * K + k] * W[(size_t)k * N + n];
    if (act == 1) s = gelu_exact(s);
    else if (act == 2) s = tanhf(s);
    Out[i] = s;
}

__global__ void rope_kernel(float* __restrict__ base, int stride)
{
    int i = blockIdx.x * blockDim.x + threadIdx.x;
    if (i >= B_ * N_ * NH_ * (HD_ / 2)) return;
    int p = i & 31; i >>= 5;
    int h = i & 15; i >>= 4;
    int n = i & (N_ - 1); i >>= 10;
    int b = i;
    float inv = __expf(-(float)(2 * p) / (float)HD_ * 9.210340371976184f);
    float ang = (float)n * inv;
    float sn, cs;
    __sincosf(ang, &sn, &cs);
    float* ptr = base + ((size_t)(b * N_ + n)) * stride + h * HD_ + 2 * p;
    float x1 = ptr[0], x2 = ptr[1];
    ptr[0] = x1 * cs - x2 * sn;
    ptr[1] = x2 * cs + x1 * sn;
}

__global__ void dn_kernel(const float* __restrict__ plucker, float* __restrict__ dn)
{
    int i = blockIdx.x * blockDim.x + threadIdx.x;
    if (i >= B_ * N_) return;
    const float* p = plucker + (size_t)i * 6;
    float x = p[0], y = p[1], z = p[2];
    float inv = 1.0f / fmaxf(sqrtf(x * x + y * y + z * z), 1e-12f);
    dn[i * 3 + 0] = x * inv;
    dn[i * 3 + 1] = y * inv;
    dn[i * 3 + 2] = z * inv;
}

__global__ void sd_kernel(const float* __restrict__ pp, float* __restrict__ sd)
{
    int i = blockIdx.x * blockDim.x + threadIdx.x;
    if (i >= B_ * N_ * 2 * PP_) return;
    int c = i & 255; int bn = i >> 8;
    int x = bn & 31, y = (bn >> 5) & 31, b = bn >> 10;
    float v = 0.0f;
    if (c < PP_) {
        if (x < 31) {
            size_t o = ((size_t)(b * N_ + y * 32 + x)) * PP_ + c;
            v = fabsf(pp[o + PP_] - pp[o]);
        }
    } else {
        if (y < 31) {
            size_t o = ((size_t)(b * N_ + y * 32 + x)) * PP_ + (c - PP_);
            v = fabsf(pp[o + 32 * PP_] - pp[o]);
        }
    }
    sd[i] = v;
}

__global__ void concat_kernel(const float* __restrict__ cop, const float* __restrict__ pp,
                              float* __restrict__ cat)
{
    int i = blockIdx.x * blockDim.x + threadIdx.x;
    const int W = C_ + PP_;
    if (i >= B_ * N_ * W) return;
    int c = i % W, bn = i / W;
    cat[i] = (c < C_) ? cop[(size_t)bn * C_ + c] : pp[(size_t)bn * PP_ + (c - C_)];
}

// ---------------------------------------------------------------------------
extern "C" void kernel_launch(void* const* d_in, const int* in_sizes, int n_in,
                              void* d_out, int out_size, void* d_ws, size_t ws_size,
                              hipStream_t stream)
{
    const float* x       = (const float*)d_in[0];
    const float* plucker = (const float*)d_in[1];
    const float* w_qkv   = (const float*)d_in[2];
    const float* w_sp    = (const float*)d_in[3];
    const float* b_sp    = (const float*)d_in[4];
    const float* pl_w1   = (const float*)d_in[5];
    const float* pl_b1   = (const float*)d_in[6];
    const float* pl_w2   = (const float*)d_in[7];
    const float* pl_b2   = (const float*)d_in[8];
    const float* pl_w3   = (const float*)d_in[9];
    const float* pl_b3   = (const float*)d_in[10];
    const float* w_cq    = (const float*)d_in[11];
    const float* w_pk    = (const float*)d_in[12];
    const float* w_pv    = (const float*)d_in[13];
    const float* w_cp    = (const float*)d_in[14];
    const float* b_cp    = (const float*)d_in[15];
    const float* rayb    = (const float*)d_in[16];
    const float* ds_w1   = (const float*)d_in[17];
    const float* ds_b1   = (const float*)d_in[18];
    const float* ds_w2   = (const float*)d_in[19];
    const float* ds_b2   = (const float*)d_in[20];
    const float* gb_w1   = (const float*)d_in[21];
    const float* gb_b1   = (const float*)d_in[22];
    const float* gb_w2   = (const float*)d_in[23];
    const float* gb_b2   = (const float*)d_in[24];
    const float* rel_t   = (const float*)d_in[25];
    const float* gfe_w1  = (const float*)d_in[26];
    const float* gfe_b1  = (const float*)d_in[27];
    const float* gfe_w2  = (const float*)d_in[28];
    const float* gfe_b2  = (const float*)d_in[29];
    float* out = (float*)d_out;
    float* ws  = (float*)d_ws;

    float* qkv = ws + 0;
    float* cq  = ws + 0;
    float* ck  = ws + 4194304;
    float* cv  = ws + 8388608;
    float* ao  = ws + 12582912;
    float* sa  = ws + 16777216;
    float* pp1 = ws + 20971520;
    float* pp2 = ws + 22020096;
    float* pp  = ws + 22544384;
    float* dsh = ws + 23068672;
    float* ds  = ws + 23330816;
    float* dnb = ws + 23396352;
    float* sd  = ws + 23408640;
    float* gbh = ws + 24457216;
    float* gb  = ws + 26554368;
    float* cop = ws + 26619904;
    float* cat = ws + 30814208;

    const int T = 256;
    auto gemm_blocks = [](int M, int N) { return (M / 128) * (N / 64); };

    wmma_gemm_kernel<<<gemm_blocks(M_, 3 * C_), T, 0, stream>>>(x, w_qkv, nullptr, nullptr, qkv, M_, C_, 3 * C_, 0);
    {
        int nb = (B_ * N_ * NH_ * (HD_ / 2) + T - 1) / T;
        rope_kernel<<<nb, T, 0, stream>>>(qkv, 3 * C_);
        rope_kernel<<<nb, T, 0, stream>>>(qkv + C_, 3 * C_);
    }
    attn_kernel<0><<<B_ * NH_ * (N_ / 128), T, 0, stream>>>(
        qkv, qkv + C_, qkv + 2 * C_, 3 * C_, 3 * C_, 3 * C_,
        rel_t, nullptr, nullptr, nullptr, nullptr, ao);
    wmma_gemm_kernel<<<gemm_blocks(M_, C_), T, 0, stream>>>(ao, w_sp, b_sp, nullptr, sa, M_, C_, C_, 0);
    small_gemm_kernel<<<(M_ * 256 + T - 1) / T, T, 0, stream>>>(plucker, pl_w1, pl_b1, pp1, M_, 6, 256, 1);
    wmma_gemm_kernel<<<gemm_blocks(M_, 128), T, 0, stream>>>(pp1, pl_w2, pl_b2, nullptr, pp2, M_, 256, 128, 1);
    wmma_gemm_kernel<<<gemm_blocks(M_, 128), T, 0, stream>>>(pp2, pl_w3, pl_b3, nullptr, pp, M_, 128, 128, 0);
    wmma_gemm_kernel<<<gemm_blocks(M_, C_), T, 0, stream>>>(sa, w_cq, nullptr, nullptr, cq, M_, C_, C_, 0);
    wmma_gemm_kernel<<<gemm_blocks(M_, C_), T, 0, stream>>>(pp, w_pk, nullptr, nullptr, ck, M_, 128, C_, 0);
    wmma_gemm_kernel<<<gemm_blocks(M_, C_), T, 0, stream>>>(pp, w_pv, nullptr, nullptr, cv, M_, 128, C_, 0);
    {
        int nb = (B_ * N_ * NH_ * (HD_ / 2) + T - 1) / T;
        rope_kernel<<<nb, T, 0, stream>>>(cq, C_);
        rope_kernel<<<nb, T, 0, stream>>>(ck, C_);
    }
    dn_kernel<<<(B_ * N_ + T - 1) / T, T, 0, stream>>>(plucker, dnb);
    small_gemm_kernel<<<(M_ * 64 + T - 1) / T, T, 0, stream>>>(plucker, ds_w1, ds_b1, dsh, M_, 6, 64, 1);
    small_gemm_kernel<<<(M_ * 16 + T - 1) / T, T, 0, stream>>>(dsh, ds_w2, ds_b2, ds, M_, 64, 16, 2);
    sd_kernel<<<(B_ * N_ * 256 + T - 1) / T, T, 0, stream>>>(pp, sd);
    wmma_gemm_kernel<<<gemm_blocks(M_, 512), T, 0, stream>>>(sd, gb_w1, gb_b1, nullptr, gbh, M_, 256, 512, 1);
    small_gemm_kernel<<<(M_ * 16 + T - 1) / T, T, 0, stream>>>(gbh, gb_w2, gb_b2, gb, M_, 512, 16, 2);
    attn_kernel<1><<<B_ * NH_ * (N_ / 128), T, 0, stream>>>(
        cq, ck, cv, C_, C_, C_,
        nullptr, dnb, ds, gb, rayb, ao);
    wmma_gemm_kernel<<<gemm_blocks(M_, C_), T, 0, stream>>>(ao, w_cp, b_cp, nullptr, cop, M_, C_, C_, 0);
    concat_kernel<<<(B_ * N_ * (C_ + PP_) + T - 1) / T, T, 0, stream>>>(cop, pp, cat);
    wmma_gemm_kernel<<<gemm_blocks(M_, C_), T, 0, stream>>>(cat, gfe_w1, gfe_b1, nullptr, ao, M_, C_ + PP_, C_, 1);
    wmma_gemm_kernel<<<gemm_blocks(M_, C_), T, 0, stream>>>(ao, gfe_w2, gfe_b2, x, out, M_, C_, C_, 0);
}
